// TestGradient_20151986553164
// MI455X (gfx1250) — matrix operand order repaired and checked by reading, not compiled, not run on hardware
//
#include <hip/hip_runtime.h>

typedef __attribute__((ext_vector_type(16))) _Float16 v16h;
typedef __attribute__((ext_vector_type(8)))  _Float16 v8h;
typedef __attribute__((ext_vector_type(4)))  _Float16 v4h;
typedef __attribute__((ext_vector_type(8)))  float    v8f;

#define BATCH 1536
#define DIN   1536
#define DOUT  4608
#define BSZ   128
#define NI    12            // DIN / BSZ
#define NJ    36            // DOUT / BSZ
#define NJP   48            // NJ padded to multiple of 16 (3 N-tiles)
#define MT    (BATCH / 16)  // 96 M-tiles

// workspace layout (bytes)
#define WS_ACC_OFF   0
#define WS_XH_OFF    1024
#define WS_WSUM_OFF  (1024 + (size_t)BATCH * DIN * 2)

// ---------------------------------------------------------------- zero scalar
__global__ void k_zero(float* acc) { *acc = 0.0f; }

// --------------------------------------------- Wsum[jb][i] = sum_{j in jb} W[i][j]
// one wave per (i, jb); lane loads float4 of the 128-wide block, wave-reduce.
__global__ void k_wsum(const float* __restrict__ W, _Float16* __restrict__ WsumH) {
  const int wid  = (blockIdx.x * blockDim.x + threadIdx.x) >> 5;
  const int lane = threadIdx.x & 31;
  const int i  = wid / NJP;
  const int jb = wid % NJP;
  if (i >= DIN) return;
  float s = 0.0f;
  if (jb < NJ) {
    const float4* p = (const float4*)(W + (size_t)i * DOUT + (size_t)jb * BSZ);
    float4 v = p[lane];
    s = v.x + v.y + v.z + v.w;
#pragma unroll
    for (int off = 16; off; off >>= 1) s += __shfl_xor(s, off);
  }
  // stored transposed ([jb][i]) so the WMMA B-tile load is contiguous per lane
  if (lane == 0) WsumH[(size_t)jb * DIN + i] = (_Float16)s;
}

// ---------------------------------------------------------- X f32 -> f16 copy
__global__ void k_xcvt(const float* __restrict__ X, _Float16* __restrict__ Xh) {
  size_t idx = ((size_t)blockIdx.x * blockDim.x + threadIdx.x) * 4;
  float4 v = *(const float4*)(X + idx);
  v4h h = { (_Float16)v.x, (_Float16)v.y, (_Float16)v.z, (_Float16)v.w };
  *(v4h*)(Xh + idx) = h;
}

// ------------------------------------------- gated tile GEMM + global reduce
// one wave per (m_tile, n_tile, ib): C[16x16] = X_tile[16x128] * Wsum_tile[128x16]
// via 4x v_wmma_f32_16x16x32_f16, then gate by G and reduce to a scalar.
__global__ void k_gemm(const _Float16* __restrict__ Xh,
                       const _Float16* __restrict__ WsumH,
                       const float* __restrict__ G,
                       float* __restrict__ acc) {
  const int lane = threadIdx.x & 31;
  const int wid  = threadIdx.x >> 5;
  const int tile = blockIdx.x * 4 + wid;     // blockDim.x == 128 -> 4 waves

  const int m_tile = tile / (3 * NI);
  const int rem    = tile % (3 * NI);
  const int n_tile = rem / NI;
  const int ib     = rem % NI;

  const int m0 = m_tile * 16;
  const int n0 = n_tile * 16;
  const int h  = lane >> 4;     // K-half selector (ISA A/B layouts)
  const int ln = lane & 15;     // A: M row; B: N column

  v8f c = {};
  const _Float16* arow = Xh    + (size_t)(m0 + ln) * DIN + (size_t)ib * BSZ;
  const _Float16* bcol = WsumH + (size_t)(n0 + ln) * DIN + (size_t)ib * BSZ + 16 * h;

#pragma unroll
  for (int kk = 0; kk < 4; ++kk) {
    // A-matrix 16x32: lanes 0-15 hold K = 0..15, lanes 16-31 hold K = 16..31
    // (contiguous, matching B)
    const _Float16* ap = arow + kk * 32;
    v16h a = *(const v16h*)(ap + 16 * h);
    // B-matrix 32x16: lane holds 16 contiguous K values for its column
    v16h b = *(const v16h*)(bcol + kk * 32);
    c = __builtin_amdgcn_wmma_f32_16x16x32_f16(false, a, false, b,
                                               (short)0, c, false, false);
  }

  // C/D layout: lanes 0-15 -> N=lane, M=v; lanes 16-31 -> N=lane-16, M=8+v
  float part = 0.0f;
  const int jb = n0 + ln;
  if (jb < NJ) {   // padded Wsum columns (36..47) carry zeros; skip their G
    const float* g = G + (size_t)(m0 + 8 * h) * (NI * NJ) + (size_t)ib * NJ + jb;
#pragma unroll
    for (int v = 0; v < 8; ++v) part += c[v] * g[(size_t)v * (NI * NJ)];
  }
#pragma unroll
  for (int off = 16; off; off >>= 1) part += __shfl_xor(part, off);
  if (lane == 0) atomicAdd(acc, part);
}

// ------------------------------------------------------------------ finalize
__global__ void k_final(const float* __restrict__ acc, float* __restrict__ out) {
  float s = *acc;
  out[0] = s * s;
}

extern "C" void kernel_launch(void* const* d_in, const int* in_sizes, int n_in,
                              void* d_out, int out_size, void* d_ws, size_t ws_size,
                              hipStream_t stream) {
  (void)in_sizes; (void)n_in; (void)out_size; (void)ws_size;
  const float* X = (const float*)d_in[0];
  const float* W = (const float*)d_in[1];
  const float* G = (const float*)d_in[2];
  float* out = (float*)d_out;

  char* ws = (char*)d_ws;
  float*    acc   = (float*)(ws + WS_ACC_OFF);
  _Float16* Xh    = (_Float16*)(ws + WS_XH_OFF);
  _Float16* WsumH = (_Float16*)(ws + WS_WSUM_OFF);

  hipLaunchKernelGGL(k_zero, dim3(1), dim3(1), 0, stream, acc);
  // 1536*48 waves, 8 waves per 256-thread block
  hipLaunchKernelGGL(k_wsum, dim3((DIN * NJP) / 8), dim3(256), 0, stream, W, WsumH);
  // 1536*1536 elements, 4 per thread
  hipLaunchKernelGGL(k_xcvt, dim3((BATCH * DIN) / (4 * 256)), dim3(256), 0, stream, X, Xh);
  // 96 * 3 * 12 = 3456 tiles, 4 waves per 128-thread block
  hipLaunchKernelGGL(k_gemm, dim3((MT * 3 * NI) / 4), dim3(128), 0, stream,
                     Xh, WsumH, G, acc);
  hipLaunchKernelGGL(k_final, dim3(1), dim3(1), 0, stream, acc, out);
}
